// MultiHeadSelfAttention_58222576665151
// MI455X (gfx1250) — matrix operand order repaired and checked by reading, not compiled, not run on hardware
//
#include <hip/hip_runtime.h>

// ---------------------------------------------------------------- types ----
typedef __attribute__((ext_vector_type(16))) __bf16 v16bf;
typedef __attribute__((ext_vector_type(8)))  __bf16 v8bf;
typedef __attribute__((ext_vector_type(8)))  float  v8f;
typedef __attribute__((ext_vector_type(4)))  int    v4i;

union V16U { v16bf v; v8bf h[2]; };

// f32 -> bf16 bits, round-to-nearest-even
__device__ __forceinline__ unsigned short f2bf(float f) {
    unsigned int u = __float_as_uint(f);
    unsigned int r = u + 0x7fffu + ((u >> 16) & 1u);
    return (unsigned short)(r >> 16);
}

// A fragment: 16x32 bf16, row-major source, leading dim ld (elements).
// Lane l: row = l&15; lanes 0-15 hold K = 0..15 ; lanes 16-31 hold K = 16..31.
__device__ __forceinline__ v16bf load_a_frag(const unsigned short* base, int ld, int lane) {
    int r = lane & 15, ks = lane >> 4;
    const unsigned short* p = base + (size_t)r * ld + ks * 16;
    V16U u;
    u.h[0] = *(const v8bf*)(p);
    u.h[1] = *(const v8bf*)(p + 8);
    return u.v;
}

// B fragment: 32x16 (KxN) sourced from row-major [N][K] storage (transposed weights),
// leading dim ld. Lane l: n = l&15; lanes 0-15 hold K=0..15, lanes 16-31 hold K=16..31.
__device__ __forceinline__ v16bf load_b_frag(const unsigned short* base, int ld, int lane) {
    int n = lane & 15, ks = lane >> 4;
    const unsigned short* p = base + (size_t)n * ld + ks * 16;
    V16U u;
    u.h[0] = *(const v8bf*)(p);
    u.h[1] = *(const v8bf*)(p + 8);
    return u.v;
}

#define WMMA_BF16(A, B, C) \
    __builtin_amdgcn_wmma_f32_16x16x32_bf16(false, (A), false, (B), (short)0, (C), false, false)

// 16-byte global -> LDS async copy (ASYNCcnt-tracked on CDNA5); guarded fallback.
__device__ __forceinline__ void async_copy_b128(const unsigned short* g, unsigned short* l) {
#if __has_builtin(__builtin_amdgcn_global_load_async_to_lds_b128)
    __builtin_amdgcn_global_load_async_to_lds_b128(
        (__attribute__((address_space(1))) v4i*)g,
        (__attribute__((address_space(3))) v4i*)l, 0, 0);
#else
    *(v8bf*)l = *(const v8bf*)g;    // sync fallback: global_load_b128 + ds_store_b128
#endif
}

__device__ __forceinline__ void wait_async0() {
#if __has_builtin(__builtin_amdgcn_s_wait_asynccnt)
    __builtin_amdgcn_s_wait_asynccnt(0);
#else
    __asm__ volatile("s_wait_asynccnt 0x0" ::: "memory");
#endif
}

// ------------------------------------------------------------ prep kernels ----
__global__ __launch_bounds__(256) void cvt_f32_bf16(const float* __restrict__ in,
                                                    unsigned short* __restrict__ out,
                                                    int nelem4) {
    int i = blockIdx.x * 256 + threadIdx.x;
    if (i < nelem4) {
        float4 f = *(const float4*)(in + (size_t)i * 4);
        ushort4 o;
        o.x = f2bf(f.x); o.y = f2bf(f.y); o.z = f2bf(f.z); o.w = f2bf(f.w);
        *(ushort4*)(out + (size_t)i * 4) = o;
    }
}

__global__ __launch_bounds__(256) void transpose_cvt(const float* __restrict__ in,
                                                     unsigned short* __restrict__ out,
                                                     int R, int C) {
    size_t i = (size_t)blockIdx.x * 256 + threadIdx.x;
    if (i < (size_t)R * C) {
        int r = (int)(i / C), c = (int)(i % C);
        out[(size_t)c * R + r] = f2bf(in[i]);
    }
}

// ------------------------------------------------------------ QKV GEMM ----
// A: Xb [4096][1024] bf16   B: WqkvT [3072][1024] bf16 (transposed, so K contiguous)
// Scatters Q (scaled 1/8), K into [b][h][t][64], V transposed into [b][h][64][t].
__global__ __launch_bounds__(256) void qkv_gemm(const unsigned short* __restrict__ A,
                                                const unsigned short* __restrict__ BT,
                                                unsigned short* __restrict__ Q,
                                                unsigned short* __restrict__ K,
                                                unsigned short* __restrict__ VT) {
    int lane = threadIdx.x & 31;
    int wave = threadIdx.x >> 5;                       // 0..7
    int row0 = blockIdx.y * 32 + (wave >> 2) * 16;     // 4096 rows
    int col0 = blockIdx.x * 64 + (wave & 3) * 16;      // 3072 cols
    const unsigned short* a0 = A + (size_t)row0 * 1024;
    const unsigned short* b0 = BT + (size_t)col0 * 1024;
    v8f acc = {};
#pragma unroll 4
    for (int kk = 0; kk < 1024; kk += 32) {
        v16bf af = load_a_frag(a0 + kk, 1024, lane);
        v16bf bf = load_b_frag(b0 + kk, 1024, lane);
        acc = WMMA_BF16(af, bf, acc);
    }
    int n = lane & 15, mh = (lane >> 4) * 8;
    int c = col0 + n;
    int which = c >> 10;           // 0=q 1=k 2=v
    int cd = c & 1023;
    int h = cd >> 6, dh = cd & 63;
#pragma unroll
    for (int vg = 0; vg < 8; ++vg) {
        int row = row0 + mh + vg;
        int b = row >> 11, t = row & 2047;
        float val = acc[vg];
        size_t rowmaj = ((size_t)(b * 16 + h) * 2048 + t) * 64 + dh;
        if (which == 0)      Q[rowmaj] = f2bf(val * 0.125f);       // fold 1/sqrt(64)
        else if (which == 1) K[rowmaj] = f2bf(val);
        else VT[((size_t)(b * 16 + h) * 64 + dh) * 2048 + t] = f2bf(val);
    }
}

// ------------------------------------------------------- flash attention ----
// Block: 4 waves, 64 q-rows. K/V tiles (shared by all 4 waves) are staged into
// LDS with async global->LDS copies each k-step; P goes through per-wave LDS to
// convert C-layout -> A-layout for the P@V WMMA.
__global__ __launch_bounds__(128) void flash_attn(const unsigned short* __restrict__ Qb,
                                                  const unsigned short* __restrict__ Kb,
                                                  const unsigned short* __restrict__ VTb,
                                                  const float* __restrict__ bias,
                                                  const int* __restrict__ mask,
                                                  unsigned short* __restrict__ CT) {
    __shared__ unsigned short sK[32 * 64];      // K tile: rows k0..k0+31, 64 dh   (4 KB)
    __shared__ unsigned short sV[64 * 32];      // V^T tile: rows dh 0..63, 32 k   (4 KB)
    __shared__ unsigned short lP[4][16 * 32];   // per-wave P staging              (4 KB)

    int tid  = threadIdx.x;
    int lane = tid & 31;
    int wave = tid >> 5;
    int h = blockIdx.y, b = blockIdx.z;
    int q0 = blockIdx.x * 64 + wave * 16;       // local q row within [0,2048)

    const unsigned short* Qp = Qb + (size_t)(b * 16 + h) * 2048 * 64;
    const unsigned short* Kp = Kb + (size_t)(b * 16 + h) * 2048 * 64;
    const unsigned short* Vp = VTb + (size_t)(b * 16 + h) * 64 * 2048;
    const float* biasp = bias + ((size_t)h * 2048 + q0) * 2048;
    const int*   maskp = mask + ((size_t)b * 2048 + q0) * 2048;

    // Q A-fragments are loop-invariant: hoist (DH=64 -> two K=32 fragments)
    v16bf qa0 = load_a_frag(Qp + (size_t)q0 * 64,      64, lane);
    v16bf qa1 = load_a_frag(Qp + (size_t)q0 * 64 + 32, 64, lane);

    float rmax[8], rsum[8];
    v8f accO[4] = {};
#pragma unroll
    for (int i = 0; i < 8; ++i) { rmax[i] = -1e30f; rsum[i] = 0.f; }

    int n = lane & 15, mh = (lane >> 4) * 8;
    unsigned short* lp = lP[wave];

    for (int k0 = 0; k0 < 2048; k0 += 32) {
        // ---- cooperative async stage of K (contiguous 4 KB) and V^T (64 rows x 64 B)
        // 256 chunks of 16 B each per tile; 128 threads -> 2 chunks each.
#pragma unroll
        for (int j = 0; j < 2; ++j) {
            int c = tid + 128 * j;                       // 0..255
            async_copy_b128(Kp + (size_t)k0 * 64 + c * 8, sK + c * 8);
            int d = c >> 2, cc = c & 3;                  // V: 4 chunks per dh row
            async_copy_b128(Vp + (size_t)d * 2048 + k0 + cc * 8, sV + d * 32 + cc * 8);
        }
        wait_async0();
        __syncthreads();

        // ---- S = (Q/sqrt(d)) K^T : two 16-wide k tiles, K-dim 64 = 2 WMMAs each
        v8f s[2] = {};
#pragma unroll
        for (int kt = 0; kt < 2; ++kt) {
            const unsigned short* kb = sK + kt * 16 * 64;
            v16bf b0f = load_b_frag(kb,      64, lane);
            v16bf b1f = load_b_frag(kb + 32, 64, lane);
            s[kt] = WMMA_BF16(qa0, b0f, s[kt]);
            s[kt] = WMMA_BF16(qa1, b1f, s[kt]);
        }
        // ---- bias + mask (C-layout: lane holds (m = mh+vg, n))
#pragma unroll
        for (int kt = 0; kt < 2; ++kt)
#pragma unroll
            for (int vg = 0; vg < 8; ++vg) {
                int m = mh + vg;
                size_t kc = (size_t)k0 + kt * 16 + n;
                float sv = s[kt][vg] + biasp[(size_t)m * 2048 + kc];
                s[kt][vg] = (maskp[(size_t)m * 2048 + kc] == 0) ? -INFINITY : sv;
            }
        if (k0 + 32 < 2048)   // stream next bias rows (dominant HBM traffic)
            __builtin_prefetch(biasp + (size_t)mh * 2048 + k0 + 32, 0, 1);

        // ---- online softmax across the 32 new columns
        float scale[8];
#pragma unroll
        for (int vg = 0; vg < 8; ++vg) {
            float tm = fmaxf(s[0][vg], s[1][vg]);
            tm = fmaxf(tm, __shfl_xor(tm, 1, 16));
            tm = fmaxf(tm, __shfl_xor(tm, 2, 16));
            tm = fmaxf(tm, __shfl_xor(tm, 4, 16));
            tm = fmaxf(tm, __shfl_xor(tm, 8, 16));
            float mnew = fmaxf(rmax[vg], tm);
            float sc = __expf(rmax[vg] - mnew);
            float p0 = __expf(s[0][vg] - mnew);
            float p1 = __expf(s[1][vg] - mnew);
            s[0][vg] = p0; s[1][vg] = p1;
            float ts = p0 + p1;
            ts += __shfl_xor(ts, 1, 16);
            ts += __shfl_xor(ts, 2, 16);
            ts += __shfl_xor(ts, 4, 16);
            ts += __shfl_xor(ts, 8, 16);
            rsum[vg] = rsum[vg] * sc + ts;
            rmax[vg] = mnew;
            scale[vg] = sc;
        }
#pragma unroll
        for (int dt = 0; dt < 4; ++dt)
#pragma unroll
            for (int vg = 0; vg < 8; ++vg) accO[dt][vg] *= scale[vg];

        // ---- P (C-layout) -> LDS -> A-fragment (per-wave private region)
#pragma unroll
        for (int kt = 0; kt < 2; ++kt)
#pragma unroll
            for (int vg = 0; vg < 8; ++vg)
                lp[(mh + vg) * 32 + kt * 16 + n] = f2bf(s[kt][vg]);
        __asm__ volatile("s_wait_dscnt 0x0" ::: "memory");
        v16bf pa = load_a_frag(lp, 32, lane);

        // ---- O += P(16x32) @ V(32x16) for 4 d-tiles (sV row-major [dh][k_local])
#pragma unroll
        for (int dt = 0; dt < 4; ++dt) {
            v16bf vb = load_b_frag(sV + dt * 16 * 32, 32, lane);
            accO[dt] = WMMA_BF16(pa, vb, accO[dt]);
        }
        __syncthreads();   // protect sK/sV before next iteration's async stage
    }

    // ---- finalize: O /= rowsum, write context bf16 [b][t][h*64+dh]
    unsigned short* ct = CT + (size_t)b * 2048 * 1024 + (size_t)h * 64;
#pragma unroll
    for (int dt = 0; dt < 4; ++dt)
#pragma unroll
        for (int vg = 0; vg < 8; ++vg) {
            int q = q0 + mh + vg;
            ct[(size_t)q * 1024 + dt * 16 + n] = f2bf(accO[dt][vg] / rsum[vg]);
        }
}

// ---------------------------------------------------------- output GEMM ----
// CT [4096][1024] bf16 @ WoutT [1024][1024] bf16 -> out f32 [4096][1024]
__global__ __launch_bounds__(256) void out_gemm(const unsigned short* __restrict__ A,
                                                const unsigned short* __restrict__ BT,
                                                float* __restrict__ out) {
    int lane = threadIdx.x & 31;
    int wave = threadIdx.x >> 5;
    int row0 = blockIdx.y * 32 + (wave >> 2) * 16;
    int col0 = blockIdx.x * 64 + (wave & 3) * 16;
    const unsigned short* a0 = A + (size_t)row0 * 1024;
    const unsigned short* b0 = BT + (size_t)col0 * 1024;
    v8f acc = {};
#pragma unroll 4
    for (int kk = 0; kk < 1024; kk += 32) {
        v16bf af = load_a_frag(a0 + kk, 1024, lane);
        v16bf bf = load_b_frag(b0 + kk, 1024, lane);
        acc = WMMA_BF16(af, bf, acc);
    }
    int n = lane & 15, mh = (lane >> 4) * 8;
#pragma unroll
    for (int vg = 0; vg < 8; ++vg)
        out[(size_t)(row0 + mh + vg) * 1024 + col0 + n] = acc[vg];
}

// -------------------------------------------------------------- launcher ----
extern "C" void kernel_launch(void* const* d_in, const int* in_sizes, int n_in,
                              void* d_out, int out_size, void* d_ws, size_t ws_size,
                              hipStream_t stream) {
    const float* x    = (const float*)d_in[0];   // (2,2048,1024)
    const int*   mask = (const int*)d_in[1];     // (2,1,2048,2048)
    const float* bias = (const float*)d_in[2];   // (16,2048,2048)
    const float* wqkv = (const float*)d_in[3];   // (1024,3072)
    const float* wout = (const float*)d_in[4];   // (1024,1024)
    float* out = (float*)d_out;                  // (2,2048,1024)

    char* ws = (char*)d_ws;
    unsigned short* Xb    = (unsigned short*)(ws);                  //  8,388,608 B
    unsigned short* WqkvT = (unsigned short*)(ws + 8388608);        //  6,291,456 B
    unsigned short* WoutT = (unsigned short*)(ws + 14680064);       //  2,097,152 B
    unsigned short* Qb    = (unsigned short*)(ws + 16777216);       //  8,388,608 B
    unsigned short* Kb    = (unsigned short*)(ws + 25165824);       //  8,388,608 B
    unsigned short* VTb   = (unsigned short*)(ws + 33554432);       //  8,388,608 B
    unsigned short* CTb   = (unsigned short*)(ws + 41943040);       //  8,388,608 B  (48 MB total)

    // 1) precision prep
    cvt_f32_bf16<<<dim3(4096), dim3(256), 0, stream>>>(x, Xb, 1048576);          // 4M elems /4
    transpose_cvt<<<dim3(12288), dim3(256), 0, stream>>>(wqkv, WqkvT, 1024, 3072);
    transpose_cvt<<<dim3(4096), dim3(256), 0, stream>>>(wout, WoutT, 1024, 1024);

    // 2) QKV projection (bf16 WMMA, f32 accum), scatters Q*0.125, K, V^T
    qkv_gemm<<<dim3(48, 128), dim3(256), 0, stream>>>(Xb, WqkvT, Qb, Kb, VTb);

    // 3) flash attention: grid (q-blocks of 64, H, B)
    flash_attn<<<dim3(32, 16, 2), dim3(128), 0, stream>>>(Qb, Kb, VTb, bias, mask, CTb);

    // 4) output projection -> f32
    out_gemm<<<dim3(16, 128), dim3(256), 0, stream>>>(CTb, WoutT, out);
}
